// CausalSelfAttention_5325759447685
// MI455X (gfx1250) — matrix operand order repaired and checked by reading, not compiled, not run on hardware
//
#include <hip/hip_runtime.h>
#include <hip/hip_bf16.h>

typedef _Float16 h_t;
typedef __attribute__((ext_vector_type(4)))  _Float16 h4;
typedef __attribute__((ext_vector_type(8)))  _Float16 h8;
typedef __attribute__((ext_vector_type(16))) _Float16 v16h;
typedef __attribute__((ext_vector_type(8)))  float    v8f;
typedef __attribute__((ext_vector_type(4)))  float    f4;

static constexpr int Bsz = 4, T = 2048, C = 1024, NH = 16, HS = 64;

__device__ __forceinline__ v16h comb(h8 lo, h8 hi) {
  return __builtin_shufflevector(lo, hi, 0,1,2,3,4,5,6,7,8,9,10,11,12,13,14,15);
}
#define WMMA_F16(a, b, c) \
  __builtin_amdgcn_wmma_f32_16x16x32_f16(false, (a), false, (b), (short)0, (c), false, false)

// ---------------------------------------------------------------------------
// Generic GEMM: out[M,N] = A[M,K] * W[K,N] + bias[N]
// MODE 0: A is fp32 (x), epilogue scatters f16 q/k/v head-major  (N = 3C)
// MODE 1: A is f16 (attention output y), epilogue stores fp32 out (N = C)
// Block tile 128x128x64, 8 waves, each wave computes 64x32 (4x2 WMMA tiles).
// ---------------------------------------------------------------------------
template <int MODE>
__global__ __launch_bounds__(256) void gemm_k(const float* __restrict__ Af,
                                              const h_t* __restrict__ Ah16,
                                              const float* __restrict__ W,
                                              const float* __restrict__ bias,
                                              float* __restrict__ out,
                                              h_t* __restrict__ qo,
                                              h_t* __restrict__ ko,
                                              h_t* __restrict__ vo,
                                              int M, int N, int K) {
  constexpr int BM = 128, BN = 128, BK = 64;
  constexpr int AS = 72, BS = 72;  // padded row strides (144B: 16B-aligned, bank-staggered)
  __shared__ h_t Ash[BM * AS];     // A tile, row-major  [m][k]
  __shared__ h_t Bsh[BN * BS];     // B tile, TRANSPOSED [n][k]

  const int tid  = threadIdx.x;
  const int lane = tid & 31;
  const int wv   = tid >> 5;
  const int half = lane >> 4;
  const int lm   = lane & 15;
  const int wm   = (wv & 1) * 64;   // wave M offset in block
  const int wn   = (wv >> 1) * 32;  // wave N offset in block
  const int m0   = blockIdx.y * BM;
  const int n0   = blockIdx.x * BN;

  const v8f vz = {0.f, 0.f, 0.f, 0.f, 0.f, 0.f, 0.f, 0.f};
  v8f acc[4][2];
#pragma unroll
  for (int i = 0; i < 4; ++i)
#pragma unroll
    for (int j = 0; j < 2; ++j) acc[i][j] = vz;

  for (int k0 = 0; k0 < K; k0 += BK) {
    __syncthreads();
    // ---- stage A tile (fp32->f16 or f16 copy), 128x64 ----
#pragma unroll
    for (int p = 0; p < 8; ++p) {
      const int r = p * 16 + (tid >> 4);
      const int c = (tid & 15) * 4;
      h4 hh;
      if (MODE == 0) {
        f4 x4 = *(const f4*)(Af + (size_t)(m0 + r) * K + k0 + c);
        hh[0] = (h_t)x4[0]; hh[1] = (h_t)x4[1]; hh[2] = (h_t)x4[2]; hh[3] = (h_t)x4[3];
      } else {
        hh = *(const h4*)(Ah16 + (size_t)(m0 + r) * K + k0 + c);
      }
      *(h4*)&Ash[r * AS + c] = hh;
    }
    // ---- stage B tile transposed: Bsh[n][k] = W[k0+k][n0+n], 64x128 ----
#pragma unroll
    for (int p = 0; p < 8; ++p) {
      const int kr = p * 8 + (tid >> 5);
      const int n4 = (tid & 31) * 4;
      f4 w4 = *(const f4*)(W + (size_t)(k0 + kr) * N + n0 + n4);
#pragma unroll
      for (int i = 0; i < 4; ++i) Bsh[(n4 + i) * BS + kr] = (h_t)w4[i];
    }
    __syncthreads();

    // ---- 2 k-steps of 32, 8 WMMAs each ----
#pragma unroll
    for (int ks = 0; ks < BK; ks += 32) {
      v16h a[4], b[2];
#pragma unroll
      for (int mt = 0; mt < 4; ++mt) {
        const int base = (wm + mt * 16 + lm) * AS + ks + half * 8;
        a[mt] = comb(*(h8*)&Ash[base], *(h8*)&Ash[base + 16]);
      }
#pragma unroll
      for (int nt = 0; nt < 2; ++nt) {
        const int base = (wn + nt * 16 + lm) * BS + ks + half * 8;
        b[nt] = comb(*(h8*)&Bsh[base], *(h8*)&Bsh[base + 16]);
      }
#pragma unroll
      for (int mt = 0; mt < 4; ++mt)
#pragma unroll
        for (int nt = 0; nt < 2; ++nt) acc[mt][nt] = WMMA_F16(a[mt], b[nt], acc[mt][nt]);
    }
  }

  // ---- epilogue ----
#pragma unroll
  for (int mt = 0; mt < 4; ++mt) {
#pragma unroll
    for (int nt = 0; nt < 2; ++nt) {
      const int n  = n0 + wn + nt * 16 + lm;
      const float bv = bias[n];
#pragma unroll
      for (int r = 0; r < 8; ++r) {
        const int m = m0 + wm + mt * 16 + r + 8 * half;
        const float val = acc[mt][nt][r] + bv;
        if (MODE == 0) {
          const int sel = n >> 10;          // 0:q 1:k 2:v
          const int c   = n & 1023;
          const int h   = c >> 6;
          const int d   = c & 63;
          const int t   = m & (T - 1);
          const int bb  = m >> 11;
          h_t* dst = (sel == 0) ? qo : (sel == 1) ? ko : vo;
          dst[(((size_t)(bb * NH + h)) * T + t) * HS + d] = (h_t)val;
        } else {
          out[(size_t)m * N + n] = val;
        }
      }
    }
  }
}

// ---------------------------------------------------------------------------
// Causal flash attention. Grid: (T/128, B*NH). 8 waves; wave owns 16 q-rows.
// q,k,v head-major f16 (B,NH,T,HS). y output (B,T,C) f16 for the proj GEMM.
// Q pre-scaled by (1/sqrt(HS))*log2(e); softmax done in exp2.
// ---------------------------------------------------------------------------
__global__ __launch_bounds__(256) void attn_k(const h_t* __restrict__ q,
                                              const h_t* __restrict__ k,
                                              const h_t* __restrict__ v,
                                              h_t* __restrict__ y) {
  __shared__ h_t Ksh[64 * 72];       // K block, natural [kk][d]  (== B^T layout for S)
  __shared__ h_t Vts[64 * 72];       // V block transposed [d][kk]
  __shared__ h_t Psh[8 * 16 * 72];   // per-wave P re-layout scratch

  const int tid  = threadIdx.x;
  const int lane = tid & 31;
  const int wv   = tid >> 5;
  const int half = lane >> 4;
  const int lm   = lane & 15;

  const int q0   = blockIdx.x * 128;
  const int head = blockIdx.y;
  const int bb   = head >> 4;
  const int hh   = head & (NH - 1);
  const size_t hb = (size_t)head * T * HS;
  const int qr   = q0 + wv * 16;  // this wave's first query row

  // load Q fragments, fold in scale * log2(e)
  const h_t qscale = (h_t)(0.125f * 1.4426950408889634f);
  h8 qs = {qscale, qscale, qscale, qscale, qscale, qscale, qscale, qscale};
  v16h qa[2];
#pragma unroll
  for (int f = 0; f < 2; ++f) {
    const h_t* base = q + hb + (size_t)(qr + lm) * HS + f * 32 + half * 8;
    h8 lo = *(const h8*)base;
    h8 hi = *(const h8*)(base + 16);
    qa[f] = comb(lo * qs, hi * qs);
  }

  const v8f vz = {0.f, 0.f, 0.f, 0.f, 0.f, 0.f, 0.f, 0.f};
  v8f o[4] = {vz, vz, vz, vz};
  float m_r[8], l_r[8];
#pragma unroll
  for (int r = 0; r < 8; ++r) { m_r[r] = -__builtin_inff(); l_r[r] = 0.f; }

  const int nkb = (q0 + 128) / 64;  // causal bound for the whole block
  for (int kb = 0; kb < nkb; ++kb) {
    __syncthreads();
    // stage K natural + V transposed (64x64 f16 each)
#pragma unroll
    for (int p = 0; p < 4; ++p) {
      const int row = p * 16 + (tid >> 4);
      const int c4  = (tid & 15) * 4;
      h4 k4 = *(const h4*)(k + hb + (size_t)(kb * 64 + row) * HS + c4);
      *(h4*)&Ksh[row * 72 + c4] = k4;
      h4 v4 = *(const h4*)(v + hb + (size_t)(kb * 64 + row) * HS + c4);
#pragma unroll
      for (int i = 0; i < 4; ++i) Vts[(c4 + i) * 72 + row] = v4[i];
    }
    __syncthreads();

    // S = Q * K^T  (16 x 64), scaled
    v8f s[4] = {vz, vz, vz, vz};
#pragma unroll
    for (int f = 0; f < 2; ++f) {
#pragma unroll
      for (int nt = 0; nt < 4; ++nt) {
        const int base = (nt * 16 + lm) * 72 + f * 32 + half * 8;
        v16h bf = comb(*(h8*)&Ksh[base], *(h8*)&Ksh[base + 16]);
        s[nt] = WMMA_F16(qa[f], bf, s[nt]);
      }
    }

    // causal mask
#pragma unroll
    for (int nt = 0; nt < 4; ++nt) {
      const int col = kb * 64 + nt * 16 + lm;
#pragma unroll
      for (int r = 0; r < 8; ++r) {
        const int row = qr + r + 8 * half;
        if (col > row) s[nt][r] = -__builtin_inff();
      }
    }

    // online softmax (row r lives in one 16-lane half; exact in wave32)
#pragma unroll
    for (int r = 0; r < 8; ++r) {
      float mx = fmaxf(fmaxf(s[0][r], s[1][r]), fmaxf(s[2][r], s[3][r]));
#pragma unroll
      for (int off = 1; off < 16; off <<= 1) mx = fmaxf(mx, __shfl_xor(mx, off, 32));
      const float mn = fmaxf(m_r[r], mx);
      const float al = exp2f(m_r[r] - mn);
      m_r[r] = mn;
      float rs = 0.f;
#pragma unroll
      for (int nt = 0; nt < 4; ++nt) {
        const float e = exp2f(s[nt][r] - mn);
        s[nt][r] = e;
        rs += e;
      }
#pragma unroll
      for (int off = 1; off < 16; off <<= 1) rs += __shfl_xor(rs, off, 32);
      l_r[r] = l_r[r] * al + rs;
#pragma unroll
      for (int dt = 0; dt < 4; ++dt) o[dt][r] *= al;
    }

    // re-layout P (C-layout -> A-layout) through per-wave LDS patch
    h_t* pw = &Psh[wv * 16 * 72];
#pragma unroll
    for (int nt = 0; nt < 4; ++nt)
#pragma unroll
      for (int r = 0; r < 8; ++r)
        pw[(r + 8 * half) * 72 + nt * 16 + lm] = (h_t)s[nt][r];

    v16h pa[2];
#pragma unroll
    for (int f = 0; f < 2; ++f) {
      const int base = lm * 72 + f * 32 + half * 8;
      pa[f] = comb(*(h8*)&pw[base], *(h8*)&pw[base + 16]);
    }

    // O += P * V
#pragma unroll
    for (int f = 0; f < 2; ++f) {
#pragma unroll
      for (int dt = 0; dt < 4; ++dt) {
        const int base = (dt * 16 + lm) * 72 + f * 32 + half * 8;
        v16h bf = comb(*(h8*)&Vts[base], *(h8*)&Vts[base + 16]);
        o[dt] = WMMA_F16(pa[f], bf, o[dt]);
      }
    }
  }

  // normalize and write y (B,T,C) as f16
#pragma unroll
  for (int r = 0; r < 8; ++r) {
    const float inv = 1.f / l_r[r];
#pragma unroll
    for (int dt = 0; dt < 4; ++dt) o[dt][r] *= inv;
  }
#pragma unroll
  for (int dt = 0; dt < 4; ++dt)
#pragma unroll
    for (int r = 0; r < 8; ++r) {
      const int t = qr + r + 8 * half;
      y[((size_t)(bb * T + t)) * C + hh * HS + dt * 16 + lm] = (h_t)o[dt][r];
    }
}

// ---------------------------------------------------------------------------
extern "C" void kernel_launch(void* const* d_in, const int* in_sizes, int n_in,
                              void* d_out, int out_size, void* d_ws, size_t ws_size,
                              hipStream_t stream) {
  const float* x      = (const float*)d_in[0];
  const float* w_attn = (const float*)d_in[1];
  const float* b_attn = (const float*)d_in[2];
  const float* w_proj = (const float*)d_in[3];
  const float* b_proj = (const float*)d_in[4];
  float* out = (float*)d_out;

  const size_t TH = (size_t)Bsz * NH * T * HS;  // 8M elems per tensor
  h_t* qb = (h_t*)d_ws;
  h_t* kb = qb + TH;
  h_t* vb = kb + TH;
  h_t* yb = vb + TH;

  const int M = Bsz * T;  // 8192
  // 1) QKV projection + head-major split
  gemm_k<0><<<dim3((3 * C) / 128, M / 128), 256, 0, stream>>>(
      x, nullptr, w_attn, b_attn, nullptr, qb, kb, vb, M, 3 * C, C);
  // 2) causal flash attention
  attn_k<<<dim3(T / 128, Bsz * NH), 256, 0, stream>>>(qb, kb, vb, yb);
  // 3) output projection
  gemm_k<1><<<dim3(C / 128, M / 128), 256, 0, stream>>>(
      nullptr, yb, w_proj, b_proj, out, nullptr, nullptr, nullptr, M, C, C);
}
